// ContrastiveTransMatcherLoss_54228257079750
// MI455X (gfx1250) — compile-verified, matrix operand order checked by reading, not run on hardware
//
#include <hip/hip_runtime.h>

typedef __attribute__((ext_vector_type(16))) _Float16 v16h;
typedef __attribute__((ext_vector_type(4)))  _Float16 v4h;
typedef __attribute__((ext_vector_type(8)))  float    v8f;

#define NB 1024          // batch
#define ND 32768         // feature dim (256*16*8)
#define TEMP_INV 10.0f   // 1/0.1
#define EPSN 1e-8f
#define TILE 128         // output tile (M=N)
#define BK 32            // k-step per WMMA stage
#define SPLITK 4
#define KCHUNK (ND / SPLITK)
#define LSTR 40          // LDS row stride in halfs (32 + 8 pad -> 80B rows, conflict-free b128)

union UFrag { float4 q[2]; v16h h; };

// ---- block-wide reductions (256 threads), deterministic order ----
__device__ __forceinline__ float redsum(float* sm, int t, float v) {
  __syncthreads();
  sm[t] = v; __syncthreads();
  #pragma unroll
  for (int o = 128; o > 0; o >>= 1) { if (t < o) sm[t] += sm[t + o]; __syncthreads(); }
  return sm[0];
}
__device__ __forceinline__ float redmax(float* sm, int t, float v) {
  __syncthreads();
  sm[t] = v; __syncthreads();
  #pragma unroll
  for (int o = 128; o > 0; o >>= 1) { if (t < o) sm[t] = fmaxf(sm[t], sm[t + o]); __syncthreads(); }
  return sm[0];
}

__device__ __forceinline__ v4h cvt4(const float4 v) {
  v4h h;
  h.x = (_Float16)v.x; h.y = (_Float16)v.y;
  h.z = (_Float16)v.z; h.w = (_Float16)v.w;
  return h;
}

// ---- Stage 1: rnorm[i] = 1 / max(||f_i||, eps) ----
__global__ __launch_bounds__(256) void ctml_norm(const float* __restrict__ f,
                                                 float* __restrict__ rnorm) {
  __shared__ float sm[256];
  const int row = blockIdx.x;
  const float4* fr = (const float4*)(f + (size_t)row * ND);
  float s = 0.f;
  for (int i = threadIdx.x; i < ND / 4; i += 256) {
    const float4 v = fr[i];
    s += v.x * v.x + v.y * v.y + v.z * v.z + v.w * v.w;
  }
  const float tot = redsum(sm, threadIdx.x, s);
  if (threadIdx.x == 0) rnorm[row] = 1.0f / fmaxf(sqrtf(tot), EPSN);
}

// ---- Stage 2: raw gram partials via f16 WMMA, f32 accumulate ----
// grid (NB/TILE, NB/TILE, SPLITK), 256 threads = 8 wave32s.
// Wave tiling: wave w owns a 32x64 sub-tile (wr=w&3 -> rows wr*32, wc=w>>2 -> cols wc*64):
// 2 A-frags x 4 B-frags = 8 WMMAs from only 12 ds_load_b128 per k-step.
// LDS double-buffered: global loads for k+1 issue before the WMMA block of k,
// converted+stored after it; one barrier per k-step.
__global__ __launch_bounds__(256) void ctml_gemm(const float* __restrict__ f,
                                                 float* __restrict__ partial) {
  __shared__ __align__(16) _Float16 As[2][TILE * LSTR];
  __shared__ __align__(16) _Float16 Bs[2][TILE * LSTR];
  const int t    = threadIdx.x;
  const int wave = t >> 5;
  const int lane = t & 31;
  const int lm   = lane & 15;   // row-in-tile (A) / col (B,C)
  const int lh   = lane >> 4;   // k-half select
  const int wr   = wave & 3;    // row group: rows wr*32 .. wr*32+31
  const int wc   = wave >> 2;   // col group: cols wc*64 .. wc*64+63
  const int rowBase = blockIdx.y * TILE;
  const int colBase = blockIdx.x * TILE;
  const size_t k0   = (size_t)blockIdx.z * KCHUNK;

  // per-thread staging slots: 4 float4 chunks for A-tile rows, 4 for B-tile rows
  int ro[4], co[4];
  const float* pa[4];
  const float* pb[4];
  #pragma unroll
  for (int i = 0; i < 4; ++i) {
    const int c = t + (i << 8);   // 0..1023 chunk id
    ro[i] = c >> 3;               // 0..127 tile row
    co[i] = c & 7;                // float4 chunk within row
    pa[i] = f + (size_t)(rowBase + ro[i]) * ND + k0 + co[i] * 4;
    pb[i] = f + (size_t)(colBase + ro[i]) * ND + k0 + co[i] * 4;
  }

  v8f acc[8] = {};   // acc[s*4+u]: rows (wr*32+s*16), cols (wc*64+u*16)

  // prologue: stage k-step 0 into buffer 0
  #pragma unroll
  for (int i = 0; i < 4; ++i) {
    const float4 va = *(const float4*)pa[i];
    const float4 vb = *(const float4*)pb[i];
    pa[i] += BK; pb[i] += BK;
    *(v4h*)(&As[0][ro[i] * LSTR + co[i] * 4]) = cvt4(va);
    *(v4h*)(&Bs[0][ro[i] * LSTR + co[i] * 4]) = cvt4(vb);
  }
  __syncthreads();

  int cur = 0;
  for (int kk = 0; kk < KCHUNK; kk += BK) {
    const bool more = (kk + BK < KCHUNK);
    // issue next k-slab's global loads up front; they drain while WMMAs run
    float4 va[4], vb[4];
    if (more) {
      #pragma unroll
      for (int i = 0; i < 4; ++i) {
        va[i] = *(const float4*)pa[i];
        vb[i] = *(const float4*)pb[i];
        pa[i] += BK; pb[i] += BK;
      }
    }

    // fragment loads: 2 A + 4 B frags (12 ds_load_b128), one wait, 8 WMMAs
    const _Float16* as = As[cur];
    const _Float16* bs = Bs[cur];
    UFrag ua[2], ub[4];   // lane lm = row/col index, k chunks at lh*8 and lh*8+16
    #pragma unroll
    for (int s = 0; s < 2; ++s) {
      const int arow = wr * 32 + s * 16 + lm;
      ua[s].q[0] = *(const float4*)(&as[arow * LSTR + lh * 8]);
      ua[s].q[1] = *(const float4*)(&as[arow * LSTR + lh * 8 + 16]);
    }
    #pragma unroll
    for (int u = 0; u < 4; ++u) {
      const int brow = wc * 64 + u * 16 + lm;
      ub[u].q[0] = *(const float4*)(&bs[brow * LSTR + lh * 8]);
      ub[u].q[1] = *(const float4*)(&bs[brow * LSTR + lh * 8 + 16]);
    }
    #pragma unroll
    for (int s = 0; s < 2; ++s) {
      #pragma unroll
      for (int u = 0; u < 4; ++u) {
        acc[s * 4 + u] = __builtin_amdgcn_wmma_f32_16x16x32_f16(
            false, ua[s].h, false, ub[u].h, (short)0, acc[s * 4 + u], false, false);
      }
    }

    // convert + store next slab into the alternate buffer
    if (more) {
      _Float16* ad = As[cur ^ 1];
      _Float16* bd = Bs[cur ^ 1];
      #pragma unroll
      for (int i = 0; i < 4; ++i) {
        *(v4h*)(&ad[ro[i] * LSTR + co[i] * 4]) = cvt4(va[i]);
        *(v4h*)(&bd[ro[i] * LSTR + co[i] * 4]) = cvt4(vb[i]);
      }
    }
    __syncthreads();
    cur ^= 1;
  }

  // C/D layout: lanes 0-15 VGPR v -> M=v; lanes 16-31 VGPR v -> M=v+8; N=lane&15
  float* outp = partial + (size_t)blockIdx.z * NB * NB;
  #pragma unroll
  for (int s = 0; s < 2; ++s) {
    #pragma unroll
    for (int u = 0; u < 4; ++u) {
      const int col = colBase + wc * 64 + u * 16 + lm;
      #pragma unroll
      for (int v = 0; v < 8; ++v) {
        const int row = rowBase + wr * 32 + s * 16 + lh * 8 + v;
        outp[(size_t)row * NB + col] = acc[s * 4 + u][v];
      }
    }
  }
}

// ---- Stage 3: per-row contrastive terms ----
__global__ __launch_bounds__(256) void ctml_row(const float* __restrict__ partial,
                                                const float* __restrict__ rnorm,
                                                const int* __restrict__ labels,
                                                float* __restrict__ rloss,
                                                float* __restrict__ rvalid,
                                                float* __restrict__ rcorr) {
  __shared__ float sm[256];
  const int row = blockIdx.x;
  const int t   = threadIdx.x;
  const int li  = labels[row];
  const float rni = rnorm[row];
  float s[4]; int cls[4]; // -1 = diagonal, 1 = positive, 0 = negative
  const float NINF = -3.0e38f;
  float offmax = NINF, mpos = NINF, mneg = NINF;
  #pragma unroll
  for (int q = 0; q < 4; ++q) {
    const int j = t + (q << 8);
    float raw = 0.f;
    #pragma unroll
    for (int z = 0; z < SPLITK; ++z)
      raw += partial[(size_t)z * NB * NB + (size_t)row * NB + j];
    const float sc = raw * rni * rnorm[j] * TEMP_INV;
    s[q] = sc;
    if (j == row) { cls[q] = -1; }
    else {
      const bool same = (labels[j] == li);
      cls[q] = same ? 1 : 0;
      offmax = fmaxf(offmax, sc);
      if (same) mpos = fmaxf(mpos, sc); else mneg = fmaxf(mneg, sc);
    }
  }
  const float rowmax = redmax(sm, t, offmax);
  float ps = 0.f, ns = 0.f;
  #pragma unroll
  for (int q = 0; q < 4; ++q) {
    if (cls[q] < 0) continue;
    const float e = __expf(s[q] - rowmax);
    if (cls[q]) ps += e; else ns += e;
  }
  const float psum = redsum(sm, t, ps);
  const float nsum = redsum(sm, t, ns);
  const float mp   = redmax(sm, t, mpos);
  const float mn   = redmax(sm, t, mneg);
  if (t == 0) {
    const bool valid = (mp > -1.0e38f) && (mn > -1.0e38f);
    rloss[row]  = valid ? (__logf(nsum) - __logf(psum)) : 0.f;
    rvalid[row] = valid ? 1.f : 0.f;
    rcorr[row]  = (valid && (mp > mn)) ? 1.f : 0.f;
  }
}

// ---- Stage 4: deterministic final reduce -> (loss, accuracy) ----
__global__ __launch_bounds__(256) void ctml_final(const float* __restrict__ rloss,
                                                  const float* __restrict__ rvalid,
                                                  const float* __restrict__ rcorr,
                                                  float* __restrict__ out) {
  __shared__ float rl[256], rv[256], rc[256];
  const int t = threadIdx.x;
  float sl = 0.f, sv = 0.f, sc = 0.f;
  #pragma unroll
  for (int q = 0; q < 4; ++q) {
    const int i = t + (q << 8);
    sl += rloss[i]; sv += rvalid[i]; sc += rcorr[i];
  }
  rl[t] = sl; rv[t] = sv; rc[t] = sc; __syncthreads();
  #pragma unroll
  for (int o = 128; o > 0; o >>= 1) {
    if (t < o) { rl[t] += rl[t + o]; rv[t] += rv[t + o]; rc[t] += rc[t + o]; }
    __syncthreads();
  }
  if (t == 0) {
    const float total = rv[0];
    out[0] = (total > 0.f) ? (rl[0] / fmaxf(total, 1.f)) : 0.f;
    out[1] = (total > 0.f) ? (rc[0] / fmaxf(total, 1.f)) : 0.5f;
  }
}

extern "C" void kernel_launch(void* const* d_in, const int* in_sizes, int n_in,
                              void* d_out, int out_size, void* d_ws, size_t ws_size,
                              hipStream_t stream) {
  (void)in_sizes; (void)n_in; (void)out_size; (void)ws_size;
  const float* feat  = (const float*)d_in[0];
  const int* labels  = (const int*)d_in[1];
  float* ws      = (float*)d_ws;
  float* partial = ws;                                  // SPLITK * NB * NB f32 (16 MB)
  float* rnorm   = partial + (size_t)SPLITK * NB * NB;  // NB
  float* rloss   = rnorm + NB;                          // NB
  float* rvalid  = rloss + NB;                          // NB
  float* rcorr   = rvalid + NB;                         // NB

  ctml_norm <<<NB, 256, 0, stream>>>(feat, rnorm);
  ctml_gemm <<<dim3(NB / TILE, NB / TILE, SPLITK), 256, 0, stream>>>(feat, partial);
  ctml_row  <<<NB, 256, 0, stream>>>(partial, rnorm, labels, rloss, rvalid, rcorr);
  ctml_final<<<1, 256, 0, stream>>>(rloss, rvalid, rcorr, (float*)d_out);
}
